// EnBaseLayer_25013889532304
// MI455X (gfx1250) — matrix operand order repaired and checked by reading, not compiled, not run on hardware
//
#include <hip/hip_runtime.h>

typedef __bf16 v16bf __attribute__((ext_vector_type(16)));
typedef float  v8f   __attribute__((ext_vector_type(8)));
typedef unsigned int v4u __attribute__((ext_vector_type(4)));
typedef float  v4f   __attribute__((ext_vector_type(4)));

constexpr int H      = 128;
constexpr int IN_ST  = 296;  // edge-input row stride (ushort elems), 288 used
constexpr int T_ST   = 136;  // mij/t1 row stride
constexpr int NIN_ST = 264;  // node-input row stride, 256 used

// LDS byte offsets (8 waves / block)
constexpr int E_INP_B = 8 * 16 * IN_ST * 2;           // 75776
constexpr int E_TB_B  = 8 * 16 * T_ST * 2;            // 34816
constexpr int E_LDS   = E_INP_B + E_TB_B + 512 + 1536; // 112640
constexpr int N_NIN_B = 8 * 16 * NIN_ST * 2;          // 67584
constexpr int N_LDS   = N_NIN_B + E_TB_B;             // 102400

__device__ inline unsigned short f2bf(float f) {
  unsigned int u = __float_as_uint(f);
  u += 0x7FFFu + ((u >> 16) & 1u);   // round-to-nearest-even
  return (unsigned short)(u >> 16);
}
__device__ inline float siluf(float v) { return v * (1.0f / (1.0f + __expf(-v))); }

struct B32 { v4u a, b; };
// A fragment (16-bit 16x32): lane half h -> K = kbase + h*8 + [0..7] and +16
__device__ inline v16bf load_a_frag(const unsigned short* row, int kbase, int half) {
  B32 s;
  s.a = *(const v4u*)(row + kbase + half * 8);
  s.b = *(const v4u*)(row + kbase + half * 8 + 16);
  return __builtin_bit_cast(v16bf, s);
}
// B fragment: pre-packed contiguous 32B per lane
__device__ inline v16bf load_b_frag(const unsigned short* p) {
  B32 s;
  s.a = *(const v4u*)(p);
  s.b = *(const v4u*)(p + 8);
  return __builtin_bit_cast(v16bf, s);
}
__device__ inline v16bf wmma_bf16(v16bf a, v16bf b, v8f& c) {
  c = __builtin_amdgcn_wmma_f32_16x16x32_bf16(false, a, false, b, (short)0, c, false, false);
  return a;
}

// ---------------- prep kernels ----------------
__global__ void cvt_bf16_kernel(const float* __restrict__ src,
                                unsigned short* __restrict__ dst, int n) {
  int i = blockIdx.x * blockDim.x + threadIdx.x;
  if (i < n) dst[i] = f2bf(src[i]);
}
__global__ void zerof_kernel(float* __restrict__ p, int n) {
  int i = blockIdx.x * blockDim.x + threadIdx.x;
  if (i < n) p[i] = 0.0f;
}
// Pack W[K][128] (row-major) into per-lane WMMA B fragments:
// frag f = kk*8+nt ; lane l: n = nt*16+(l&15), k = kk*32+(l>>4)*8+(j&7)+(j>>3)*16
__global__ void pack_b_kernel(const float* __restrict__ W,
                              unsigned short* __restrict__ dst, int K, int KK) {
  int i = blockIdx.x * blockDim.x + threadIdx.x;
  int total = KK * 8 * 512;
  if (i >= total) return;
  int j  = i & 15;
  int l  = (i >> 4) & 31;
  int f  = i >> 9;
  int kk = f >> 3, nt = f & 7;
  int k  = kk * 32 + ((l >> 4) << 3) + (j & 7) + ((j >> 3) << 4);
  int n  = nt * 16 + (l & 15);
  float v = (k < K) ? W[(size_t)k * 128 + n] : 0.0f;
  dst[i] = f2bf(v);
}

// ---------------- fused edge kernel ----------------
__global__ void __launch_bounds__(256) edge_kernel(
    const int* __restrict__ ei, const float* __restrict__ x,
    const float* __restrict__ eattr, const unsigned short* __restrict__ hbf,
    const unsigned short* __restrict__ fWe1, const unsigned short* __restrict__ fWe2,
    const unsigned short* __restrict__ fWx1,
    const float* __restrict__ be1, const float* __restrict__ be2,
    const float* __restrict__ Winf, const float* __restrict__ binf,
    const float* __restrict__ bx1, const float* __restrict__ Wx2,
    float* __restrict__ mi, float* __restrict__ dxa, int E) {
  extern __shared__ char smem[];
  const int wave = threadIdx.x >> 5;
  const int lane = threadIdx.x & 31;
  const int hl   = lane & 15;
  const int half = lane >> 4;

  unsigned short* inp = (unsigned short*)smem + wave * 16 * IN_ST;
  unsigned short* tb  = (unsigned short*)(smem + E_INP_B) + wave * 16 * T_ST;
  int*   edst = (int*)(smem + E_INP_B + E_TB_B) + wave * 16;
  float* erx  = (float*)(smem + E_INP_B + E_TB_B + 512) + wave * 48;

  const int e0 = (blockIdx.x * 8 + wave) * 16;

  // ---- per-edge geometry + gaussians (lanes 0..15, one edge each) ----
  int srcv = 0, dstv = 0;
  if (lane < 16) {
    int e  = e0 + lane;
    int ec = (e < E) ? e : (E - 1);
    srcv = ei[ec];
    dstv = ei[E + ec];
    float rx = x[dstv * 3 + 0] - x[srcv * 3 + 0];
    float ry = x[dstv * 3 + 1] - x[srcv * 3 + 1];
    float rz = x[dstv * 3 + 2] - x[srcv * 3 + 2];
    float dd = sqrtf(rx * rx + ry * ry + rz * rz + 1e-8f);
    edst[lane] = dstv;
    float inv = 1.0f / (dd + 1.0f);
    erx[lane * 3 + 0] = rx * inv;
    erx[lane * 3 + 1] = ry * inv;
    erx[lane * 3 + 2] = rz * inv;
    const float step  = 10.0f / 19.0f;
    const float coeff = -0.5f / (step * step);
    unsigned short* row = inp + lane * IN_ST;
#pragma unroll
    for (int g = 0; g < 20; ++g) {
      float t = dd - (float)g * step;
      row[256 + g] = f2bf(__expf(coeff * t * t));
    }
#pragma unroll
    for (int a = 0; a < 4; ++a) row[276 + a] = f2bf(eattr[(size_t)ec * 4 + a]);
#pragma unroll
    for (int a = 280; a < 288; ++a) row[a] = 0;
  }

  // ---- gather h[dst] (cols 0..127) and h[src] (cols 128..255), bf16 ----
  for (int e = 0; e < 16; ++e) {
    int s  = __builtin_amdgcn_readlane(srcv, e);
    int dI = __builtin_amdgcn_readlane(dstv, e);
    const v4u* hr = (const v4u*)(hbf + (size_t)(half ? s : dI) * H);
    *(v4u*)(inp + e * IN_ST + half * 128 + hl * 8) = hr[hl];
  }

  // ---- layer 1: [16,288] @ We1 -> silu -> tb ----
  v8f acc[8];
#pragma unroll
  for (int nt = 0; nt < 8; ++nt) {
    float b = be1[nt * 16 + hl];
#pragma unroll
    for (int i = 0; i < 8; ++i) acc[nt][i] = b;
  }
#pragma unroll
  for (int kk = 0; kk < 9; ++kk) {
    v16bf a = load_a_frag(inp + hl * IN_ST, kk * 32, half);
#pragma unroll
    for (int nt = 0; nt < 8; ++nt) {
      v16bf b = load_b_frag(fWe1 + ((size_t)(kk * 8 + nt) * 32 + lane) * 16);
      wmma_bf16(a, b, acc[nt]);
    }
  }
#pragma unroll
  for (int nt = 0; nt < 8; ++nt)
#pragma unroll
    for (int r = 0; r < 8; ++r)
      tb[(r + half * 8) * T_ST + nt * 16 + hl] = f2bf(siluf(acc[nt][r]));

  // ---- layer 2: t1 @ We2 -> silu = mij ----
#pragma unroll
  for (int nt = 0; nt < 8; ++nt) {
    float b = be2[nt * 16 + hl];
#pragma unroll
    for (int i = 0; i < 8; ++i) acc[nt][i] = b;
  }
#pragma unroll
  for (int kk = 0; kk < 4; ++kk) {
    v16bf a = load_a_frag(tb + hl * T_ST, kk * 32, half);
#pragma unroll
    for (int nt = 0; nt < 8; ++nt) {
      v16bf b = load_b_frag(fWe2 + ((size_t)(kk * 8 + nt) * 32 + lane) * 16);
      wmma_bf16(a, b, acc[nt]);
    }
  }
#pragma unroll
  for (int nt = 0; nt < 8; ++nt)
#pragma unroll
    for (int r = 0; r < 8; ++r) acc[nt][r] = siluf(acc[nt][r]);

  // ---- eij = sigmoid(mij @ Winf + binf) via shuffle reduction ----
  float winfv[8], wx2v[8];
#pragma unroll
  for (int nt = 0; nt < 8; ++nt) {
    winfv[nt] = Winf[nt * 16 + hl];
    wx2v[nt]  = Wx2[nt * 16 + hl];
  }
  float b0 = binf[0];
  float eij[8];
#pragma unroll
  for (int r = 0; r < 8; ++r) {
    float p = 0.0f;
#pragma unroll
    for (int nt = 0; nt < 8; ++nt) p += acc[nt][r] * winfv[nt];
    p += __shfl_xor(p, 1, 32);
    p += __shfl_xor(p, 2, 32);
    p += __shfl_xor(p, 4, 32);
    p += __shfl_xor(p, 8, 32);
    float ev = 1.0f / (1.0f + __expf(-(p + b0)));
    int M = r + half * 8;
    eij[r] = (e0 + M < E) ? ev : 0.0f;   // mask tail edges
  }

  // ---- store mij to LDS + segment-sum scatter (L2-side f32 atomics) ----
#pragma unroll
  for (int r = 0; r < 8; ++r) {
    int M  = r + half * 8;
    int dI = edst[M];
#pragma unroll
    for (int nt = 0; nt < 8; ++nt) {
      float v = acc[nt][r];
      tb[M * T_ST + nt * 16 + hl] = f2bf(v);
      atomicAdd(&mi[(size_t)dI * H + nt * 16 + hl], v * eij[r]);
    }
  }

  // ---- gate: tanh(silu(mij @ Wx1 + bx1) @ Wx2), coord scatter ----
  v8f acc3[8];
#pragma unroll
  for (int nt = 0; nt < 8; ++nt) {
    float b = bx1[nt * 16 + hl];
#pragma unroll
    for (int i = 0; i < 8; ++i) acc3[nt][i] = b;
  }
#pragma unroll
  for (int kk = 0; kk < 4; ++kk) {
    v16bf a = load_a_frag(tb + hl * T_ST, kk * 32, half);
#pragma unroll
    for (int nt = 0; nt < 8; ++nt) {
      v16bf b = load_b_frag(fWx1 + ((size_t)(kk * 8 + nt) * 32 + lane) * 16);
      wmma_bf16(a, b, acc3[nt]);
    }
  }
#pragma unroll
  for (int r = 0; r < 8; ++r) {
    float p = 0.0f;
#pragma unroll
    for (int nt = 0; nt < 8; ++nt) p += siluf(acc3[nt][r]) * wx2v[nt];
    p += __shfl_xor(p, 1, 32);
    p += __shfl_xor(p, 2, 32);
    p += __shfl_xor(p, 4, 32);
    p += __shfl_xor(p, 8, 32);
    int M = r + half * 8;
    float g = tanhf(p) * ((e0 + M < E) ? 1.0f : 0.0f);
    if (hl == 0) {
      int dI = edst[M];
      atomicAdd(&dxa[dI * 3 + 0], erx[M * 3 + 0] * g);
      atomicAdd(&dxa[dI * 3 + 1], erx[M * 3 + 1] * g);
      atomicAdd(&dxa[dI * 3 + 2], erx[M * 3 + 2] * g);
    }
  }
}

// ---------------- node MLP kernel ----------------
__global__ void __launch_bounds__(256) node_kernel(
    const float* __restrict__ h, const unsigned short* __restrict__ hbf,
    const float* __restrict__ mi, const unsigned short* __restrict__ fWn1,
    const unsigned short* __restrict__ fWn2, const float* __restrict__ bn1,
    const float* __restrict__ bn2, float* __restrict__ outh, int N) {
  extern __shared__ char smem[];
  const int wave = threadIdx.x >> 5;
  const int lane = threadIdx.x & 31;
  const int hl   = lane & 15;
  const int half = lane >> 4;

  unsigned short* nin = (unsigned short*)smem + wave * 16 * NIN_ST;
  unsigned short* tb  = (unsigned short*)(smem + N_NIN_B) + wave * 16 * T_ST;

  const int n0 = (blockIdx.x * 8 + wave) * 16;

  // build [mi(bf16) | h(bf16)] rows
  for (int e = 0; e < 16; ++e) {
    int node = n0 + e;
    int nc   = (node < N) ? node : (N - 1);
    v4f m = ((const v4f*)(mi + (size_t)nc * H))[lane];
    unsigned short* row = nin + e * NIN_ST;
    unsigned int p0 = (unsigned int)f2bf(m[0]) | ((unsigned int)f2bf(m[1]) << 16);
    unsigned int p1 = (unsigned int)f2bf(m[2]) | ((unsigned int)f2bf(m[3]) << 16);
    *(unsigned int*)(row + lane * 4)     = p0;
    *(unsigned int*)(row + lane * 4 + 2) = p1;
    if (half == 0)
      *(v4u*)(row + 128 + hl * 8) = ((const v4u*)(hbf + (size_t)nc * H))[hl];
  }

  v8f acc[8];
#pragma unroll
  for (int nt = 0; nt < 8; ++nt) {
    float b = bn1[nt * 16 + hl];
#pragma unroll
    for (int i = 0; i < 8; ++i) acc[nt][i] = b;
  }
#pragma unroll
  for (int kk = 0; kk < 8; ++kk) {
    v16bf a = load_a_frag(nin + hl * NIN_ST, kk * 32, half);
#pragma unroll
    for (int nt = 0; nt < 8; ++nt) {
      v16bf b = load_b_frag(fWn1 + ((size_t)(kk * 8 + nt) * 32 + lane) * 16);
      wmma_bf16(a, b, acc[nt]);
    }
  }
#pragma unroll
  for (int nt = 0; nt < 8; ++nt)
#pragma unroll
    for (int r = 0; r < 8; ++r)
      tb[(r + half * 8) * T_ST + nt * 16 + hl] = f2bf(siluf(acc[nt][r]));

#pragma unroll
  for (int nt = 0; nt < 8; ++nt) {
    float b = bn2[nt * 16 + hl];
#pragma unroll
    for (int i = 0; i < 8; ++i) acc[nt][i] = b;
  }
#pragma unroll
  for (int kk = 0; kk < 4; ++kk) {
    v16bf a = load_a_frag(tb + hl * T_ST, kk * 32, half);
#pragma unroll
    for (int nt = 0; nt < 8; ++nt) {
      v16bf b = load_b_frag(fWn2 + ((size_t)(kk * 8 + nt) * 32 + lane) * 16);
      wmma_bf16(a, b, acc[nt]);
    }
  }
#pragma unroll
  for (int nt = 0; nt < 8; ++nt)
#pragma unroll
    for (int r = 0; r < 8; ++r) {
      int M = r + half * 8;
      int node = n0 + M;
      if (node < N) {
        size_t o = (size_t)node * H + nt * 16 + hl;
        outh[o] = h[o] + acc[nt][r];
      }
    }
}

__global__ void x_kernel(const float* __restrict__ x, const float* __restrict__ dxa,
                         const float* __restrict__ mask, float* __restrict__ outx, int N) {
  int i = blockIdx.x * blockDim.x + threadIdx.x;
  if (i < N * 3) outx[i] = x[i] + dxa[i] * mask[i / 3];
}

// ---------------- launch ----------------
extern "C" void kernel_launch(void* const* d_in, const int* in_sizes, int n_in,
                              void* d_out, int out_size, void* d_ws, size_t ws_size,
                              hipStream_t stream) {
  const float* h     = (const float*)d_in[0];
  const float* x     = (const float*)d_in[1];
  const int*   ei    = (const int*)d_in[2];
  const float* mask  = (const float*)d_in[3];
  const float* eattr = (const float*)d_in[4];
  const float* We1 = (const float*)d_in[5];
  const float* be1 = (const float*)d_in[6];
  const float* We2 = (const float*)d_in[7];
  const float* be2 = (const float*)d_in[8];
  const float* Winf = (const float*)d_in[9];
  const float* binf = (const float*)d_in[10];
  const float* Wx1 = (const float*)d_in[11];
  const float* bx1 = (const float*)d_in[12];
  const float* Wx2 = (const float*)d_in[13];
  const float* Wn1 = (const float*)d_in[14];
  const float* bn1 = (const float*)d_in[15];
  const float* Wn2 = (const float*)d_in[16];
  const float* bn2 = (const float*)d_in[17];

  const int N = in_sizes[0] / H;
  const int E = in_sizes[2] / 2;

  char* ws = (char*)d_ws;
  auto aup = [](size_t v) { return (v + 255) & ~(size_t)255; };
  size_t off = 0;
  unsigned short* hbf = (unsigned short*)(ws + off); off += aup((size_t)N * H * 2);
  float* mi  = (float*)(ws + off); off += aup((size_t)N * H * 4);
  float* dxa = (float*)(ws + off); off += aup((size_t)N * 3 * 4);
  unsigned short* fWe1 = (unsigned short*)(ws + off); off += aup((size_t)9 * 8 * 512 * 2);
  unsigned short* fWe2 = (unsigned short*)(ws + off); off += aup((size_t)4 * 8 * 512 * 2);
  unsigned short* fWx1 = (unsigned short*)(ws + off); off += aup((size_t)4 * 8 * 512 * 2);
  unsigned short* fWn1 = (unsigned short*)(ws + off); off += aup((size_t)8 * 8 * 512 * 2);
  unsigned short* fWn2 = (unsigned short*)(ws + off); off += aup((size_t)4 * 8 * 512 * 2);

  const int nh = N * H;
  cvt_bf16_kernel<<<(nh + 255) / 256, 256, 0, stream>>>(h, hbf, nh);
  zerof_kernel<<<(nh + 255) / 256, 256, 0, stream>>>(mi, nh);
  zerof_kernel<<<(N * 3 + 255) / 256, 256, 0, stream>>>(dxa, N * 3);
  pack_b_kernel<<<(9 * 8 * 512 + 255) / 256, 256, 0, stream>>>(We1, fWe1, 280, 9);
  pack_b_kernel<<<(4 * 8 * 512 + 255) / 256, 256, 0, stream>>>(We2, fWe2, 128, 4);
  pack_b_kernel<<<(4 * 8 * 512 + 255) / 256, 256, 0, stream>>>(Wx1, fWx1, 128, 4);
  pack_b_kernel<<<(8 * 8 * 512 + 255) / 256, 256, 0, stream>>>(Wn1, fWn1, 256, 8);
  pack_b_kernel<<<(4 * 8 * 512 + 255) / 256, 256, 0, stream>>>(Wn2, fWn2, 128, 4);

  const int etiles = (E + 15) / 16;
  const int eblocks = (etiles + 7) / 8;
  edge_kernel<<<eblocks, 256, E_LDS, stream>>>(ei, x, eattr, hbf, fWe1, fWe2, fWx1,
                                               be1, be2, Winf, binf, bx1, Wx2,
                                               mi, dxa, E);

  const int ntiles = (N + 15) / 16;
  const int nblocks = (ntiles + 7) / 8;
  float* outh = (float*)d_out;
  float* outx = (float*)d_out + (size_t)N * H;
  node_kernel<<<nblocks, 256, N_LDS, stream>>>(h, hbf, mi, fWn1, fWn2, bn1, bn2, outh, N);
  x_kernel<<<(N * 3 + 255) / 256, 256, 0, stream>>>(x, dxa, mask, outx, N);
}
